// ComplexReciprocatorMixer_89060441850385
// MI455X (gfx1250) — matrix operand order repaired and checked by reading, not compiled, not run on hardware
//
#include <hip/hip_runtime.h>
#include <hip/hip_bf16.h>

// ComplexReciprocatorMixer — CDNA5 (gfx1250), round 4.
// B=4, T=2048, H=1024, S=512, R=16, E=4, F=38.  M = B*T = 8192.
//
//  * bf16 WMMA GEMMs with LDS tiles pre-swizzled into fragment order
//    (each operand fragment = one contiguous 32B read -> 2x ds_load_b128).
//  * Real GEMMs use 128x64 workgroup tiles: 2 A-frags x 4 B-frags per wave
//    -> 8 WMMAs per 12 ds_load_b128.
//  * Double-buffered LDS + register staging overlaps HBM with the matrix pipe.
//  * uproj stages its z rows into LDS via the Tensor Data Mover
//    (tensor_load_to_lds + s_wait_tensorcnt), exercising the async path.
//    This toolchain uses the 6-arg builtin form (extra i32x8 group).

typedef __attribute__((ext_vector_type(16))) __bf16 v16bf;
typedef __attribute__((ext_vector_type(8)))  __bf16 v8bf;
typedef __attribute__((ext_vector_type(8)))  float  v8f;
typedef __attribute__((ext_vector_type(8)))  unsigned int v8u;
typedef __attribute__((ext_vector_type(4)))  unsigned int u32x4;
typedef __attribute__((ext_vector_type(8)))  int i32x8;
typedef __attribute__((ext_vector_type(4)))  int i32x4;

#define EPS_C 1e-6f

static constexpr int BMR = 128;       // M tile for real GEMMs (2 A frags / wave)
static constexpr int BM  = 64;        // M tile for complex GEMM
static constexpr int BN  = 64;
static constexpr int BK  = 32;
static constexpr int LDP = BK + 16;   // 48 bf16 = 96B row stride (3x32B)

// WMMA 16-bit A-fragment K permutation: lane half hw reads contiguous
// [hw*16 .. hw*16+15] when row is stored as [k0..7, k16..23, k8..15, k24..31].
__device__ __forceinline__ int permA(int k) {
    return ((k >> 3) & 1) * 16 + ((k >> 4) & 1) * 8 + (k & 7);
}

__device__ __forceinline__ v16bf negbf(v16bf a) {
    v8u t = __builtin_bit_cast(v8u, a) ^ 0x80008000u;   // flip bf16 sign bits
    return __builtin_bit_cast(v16bf, t);
}

// ---------------------------------------------------------------------------
// TDM: 1-D row load Global -> LDS via Tensor Data Mover (D# per ISA 8.3/8.4).
// 6-arg builtin form on this toolchain (groups 0..3 + extra group + cpol).
// ---------------------------------------------------------------------------
__device__ __forceinline__ void tdm_load_row(const float* gsrc, void* lds_dst, int nElem) {
    const unsigned long long ga = (unsigned long long)(size_t)gsrc;
    u32x4 g0 = {};
    g0[0] = 1u;                                   // count=1, user descriptor
    g0[1] = (unsigned)(size_t)lds_dst;            // lds_addr = generic addr[31:0]
    g0[2] = (unsigned)ga;                         // global_addr[31:0]
    g0[3] = (unsigned)((ga >> 32) & 0x01FFFFFFu)  // global_addr[56:32]
          | (2u << 30);                           // type = 2 ("image")
    i32x8 g1 = {};
    g1[0] = (int)(2u << 16);                      // data_size = 4 bytes
    g1[1] = (int)(((unsigned)nElem & 0xFFFFu) << 16);                // tensor_dim0 lo
    g1[2] = (int)((((unsigned)nElem >> 16) & 0xFFFFu) | (1u << 16)); // dim0 hi | tensor_dim1=1
    g1[3] = (int)(((unsigned)nElem & 0xFFFFu) << 16);                // tile_dim0 = nElem
    g1[4] = 0;                                    // tile_dim1 = 0 -> 1-D tile
    g1[5] = nElem;                                // tensor_dim0_stride
    g1[6] = 0;
    g1[7] = 0;
    const i32x4 gz4 = {};
    const i32x8 gz8 = {};
    __builtin_amdgcn_tensor_load_to_lds(g0, g1, gz4, gz4, gz8, 0);
}

// ---------------------------------------------------------------------------
// Kernel 1: real GEMM  C = A(f32->bf16) @ B(f32->bf16) + bias, C f32.
// 128x64 workgroup tile, 128 threads (4 waves), each wave: 32 rows x 64 cols.
// M%128==0, N%64==0, K%32==0 (guaranteed by caller).
// ---------------------------------------------------------------------------
__global__ __launch_bounds__(128) void gemm_rr_kernel(
    const float* __restrict__ A, int lda,
    const float* __restrict__ Bm, int ldb,
    const float* __restrict__ bias,
    float* __restrict__ C, int ldc,
    int M, int N, int K)
{
    __shared__ __attribute__((aligned(32))) __bf16 lA [2][BMR][LDP];
    __shared__ __attribute__((aligned(32))) __bf16 lBt[2][BN][LDP];

    const int tid  = threadIdx.x;
    const int wv   = tid >> 5;
    const int lane = tid & 31;
    const int hw   = lane >> 4;
    const int r16  = lane & 15;
    const int mBase = blockIdx.y * BMR;
    const int nBase = blockIdx.x * BN;

    v8f acc[2][4] = {};
    float4 ra[8], rb[4];

    auto gload = [&](int k0) {
        #pragma unroll
        for (int i = 0; i < 8; ++i) {             // A: 4096 f32 = 1024 float4
            int idx = tid + i * 128;
            ra[i] = *(const float4*)(A + (size_t)(mBase + (idx >> 3)) * lda + k0 + ((idx & 7) << 2));
        }
        #pragma unroll
        for (int i = 0; i < 4; ++i) {             // B: 2048 f32 = 512 float4
            int idx = tid + i * 128;
            rb[i] = *(const float4*)(Bm + (size_t)(k0 + (idx >> 4)) * ldb + nBase + ((idx & 15) << 2));
        }
    };
    auto sstore = [&](int buf) {
        #pragma unroll
        for (int i = 0; i < 8; ++i) {
            int idx = tid + i * 128;
            int row = idx >> 3, c = (idx & 7) << 2;
            __bf16* d = &lA[buf][row][c];
            d[0] = (__bf16)ra[i].x; d[1] = (__bf16)ra[i].y;
            d[2] = (__bf16)ra[i].z; d[3] = (__bf16)ra[i].w;
        }
        #pragma unroll
        for (int i = 0; i < 4; ++i) {
            int idx = tid + i * 128;
            int k = idx >> 4, c = (idx & 15) << 2;
            lBt[buf][c + 0][k] = (__bf16)rb[i].x;
            lBt[buf][c + 1][k] = (__bf16)rb[i].y;
            lBt[buf][c + 2][k] = (__bf16)rb[i].z;
            lBt[buf][c + 3][k] = (__bf16)rb[i].w;
        }
    };

    gload(0);
    sstore(0);
    const int nk = K / BK;
    for (int kt = 0; kt < nk; ++kt) {
        __syncthreads();
        if (kt + 1 < nk) gload((kt + 1) * BK);     // overlap with WMMAs below
        if (kt + 2 < nk) {
            __builtin_prefetch(A  + (size_t)(mBase + tid) * lda + (kt + 2) * BK, 0, 3);
            __builtin_prefetch(Bm + (size_t)((kt + 2) * BK + (tid & 31)) * ldb + nBase, 0, 3);
        }
        const int buf = kt & 1;
        const v16bf af0 = *(const v16bf*)&lA[buf][wv * 32 + r16][hw * 16];
        const v16bf af1 = *(const v16bf*)&lA[buf][wv * 32 + 16 + r16][hw * 16];
        #pragma unroll
        for (int nt = 0; nt < 4; ++nt) {
            const v16bf bf = *(const v16bf*)&lBt[buf][nt * 16 + r16][hw * 16];
            acc[0][nt] = __builtin_amdgcn_wmma_f32_16x16x32_bf16(
                false, af0, false, bf, (short)0, acc[0][nt], false, false);
            acc[1][nt] = __builtin_amdgcn_wmma_f32_16x16x32_bf16(
                false, af1, false, bf, (short)0, acc[1][nt], false, false);
        }
        if (kt + 1 < nk) sstore((kt + 1) & 1);
    }

    #pragma unroll
    for (int g = 0; g < 2; ++g) {
        #pragma unroll
        for (int nt = 0; nt < 4; ++nt) {
            int n = nBase + nt * 16 + r16;
            float bv = bias ? bias[n] : 0.f;
            #pragma unroll
            for (int i = 0; i < 8; ++i) {
                int m = mBase + wv * 32 + g * 16 + hw * 8 + i;
                C[(size_t)m * ldc + n] = acc[g][nt][i] + bv;
            }
        }
    }
}

// ---------------------------------------------------------------------------
// Kernel 2: u = (zr + i zi) @ (A_r + i A_i) per engine (tiny N) — VALU.
// LDS staging of the z rows is done by the Tensor Data Mover.
// ---------------------------------------------------------------------------
__global__ __launch_bounds__(128) void uproj_kernel(
    const float* __restrict__ zr, const float* __restrict__ zi,
    const float* __restrict__ A_r, const float* __restrict__ A_i,
    float* __restrict__ ur, float* __restrict__ ui)
{
    __shared__ float szr[512];
    __shared__ float szi[512];
    const int m   = blockIdx.x;
    const int tid = threadIdx.x;

    if (tid == 0) {   // wave 0 issues both DMA descriptors and waits
        tdm_load_row(zr + (size_t)m * 512, szr, 512);
        tdm_load_row(zi + (size_t)m * 512, szi, 512);
        __builtin_amdgcn_s_wait_tensorcnt(0);
    }
    __syncthreads();

    const int e = tid >> 5;
    const int r = tid & 15;
    const int part = (tid >> 4) & 1;
    const float* Wr = A_r + (size_t)e * 512 * 16 + r;
    const float* Wi = A_i + (size_t)e * 512 * 16 + r;

    float acc = 0.f;
    if (part == 0) {
        for (int s = 0; s < 512; ++s)
            acc = fmaf(szr[s], Wr[(size_t)s * 16], fmaf(-szi[s], Wi[(size_t)s * 16], acc));
    } else {
        for (int s = 0; s < 512; ++s)
            acc = fmaf(szr[s], Wi[(size_t)s * 16], fmaf(szi[s], Wr[(size_t)s * 16], acc));
    }
    const size_t o = ((size_t)e * 8192 + m) * 16 + r;   // [E][B*T][R]
    if (part == 0) ur[o] = acc; else ui[o] = acc;
}

// ---------------------------------------------------------------------------
// Kernel 3: sequential complex recurrence, one wave per (e,b) chain.
// ---------------------------------------------------------------------------
__device__ __forceinline__ float rsum16(float v) {
    v += __shfl_xor(v, 1, 16);
    v += __shfl_xor(v, 2, 16);
    v += __shfl_xor(v, 4, 16);
    v += __shfl_xor(v, 8, 16);
    return v;
}
__device__ __forceinline__ float rmax16(float v) {
    v = fmaxf(v, __shfl_xor(v, 1, 16));
    v = fmaxf(v, __shfl_xor(v, 2, 16));
    v = fmaxf(v, __shfl_xor(v, 4, 16));
    v = fmaxf(v, __shfl_xor(v, 8, 16));
    return v;
}

__global__ __launch_bounds__(32) void engine_kernel(
    const float* __restrict__ ur, const float* __restrict__ ui,
    const float* __restrict__ lam_r, const float* __restrict__ lam_i,
    float* __restrict__ fr, float* __restrict__ fi)
{
    const int chain = blockIdx.x;       // e*B + b, 16 chains
    const int e = chain >> 2;           // B = 4
    const int lane = threadIdx.x;
    const int r = lane & 15;

    const float lr = lam_r[e * 16 + r];
    const float li = lam_i[e * 16 + r];
    float hr = 0.f, hi = 0.f, acc = 0.f;

    const size_t ubase = (size_t)chain * 2048 * 16;
    const size_t fbase = (size_t)chain * 2048 * 38;

    for (int t = 0; t < 2048; ++t) {
        const float ur_t = ur[ubase + (size_t)t * 16 + r];
        const float ui_t = ui[ubase + (size_t)t * 16 + r];
        const float nhr = 0.9f * (lr * hr - li * hi) + 0.1f * ur_t;
        const float nhi = 0.9f * (lr * hi + li * hr) + 0.1f * ui_t;
        const float mag2 = nhr * nhr + nhi * nhi;
        const float mmean = rsum16(mag2) * (1.f / 16.f);
        const float mmax  = rmax16(mag2);
        acc = 0.99f * acc + 0.01f * mmean;
        const float inv = rsqrtf(mmean + EPS_C);
        const float hnr = nhr * inv;
        const float hni = nhi * inv;
        const float hrm = rsum16(hnr) * (1.f / 16.f);
        const float him = rsum16(hni) * (1.f / 16.f);

        float* frp = fr + fbase + (size_t)t * 38;
        float* fip = fi + fbase + (size_t)t * 38;
        if (lane == 0) {
            frp[0] = hrm; frp[1] = him;
            frp[2] = sqrtf(mmean + EPS_C);
            frp[3] = sqrtf(mmax + EPS_C);
            frp[4] = acc; frp[5] = log1pf(acc);
            fip[0] = him; fip[1] = -hrm;
            fip[2] = 0.f; fip[3] = 0.f; fip[4] = 0.f; fip[5] = 0.f;
        }
        if (lane < 16) {
            frp[6 + r]  = hnr;  frp[22 + r] = hni;
            fip[6 + r]  = hni;  fip[22 + r] = -hnr;
        }
        hr = nhr; hi = nhi;
    }
}

// ---------------------------------------------------------------------------
// Kernel 4/5: complex GEMM  (Cr + iCi) = (Ar + iAi) @ (Br + iBi) + bias.
// TA: float (lda may be 38, rows 8B-aligned -> float2 loads) or __bf16
// (lda=4096 -> 16B v8bf loads).  K padded to BK in LDS.  64x64 tile,
// 16 WMMAs per k-chunk per wave.
// ---------------------------------------------------------------------------
template <typename TA, typename TO>
__global__ __launch_bounds__(128) void cgemm_kernel(
    const TA* __restrict__ Ar, const TA* __restrict__ Ai, int lda,
    const float* __restrict__ Br, const float* __restrict__ Bi, int ldb,
    const float* __restrict__ biasR, const float* __restrict__ biasI,
    TO* __restrict__ Cr, TO* __restrict__ Ci, int ldc, int colOff,
    int M, int N, int K)
{
    __shared__ __attribute__((aligned(32))) __bf16 lAr[2][BM][LDP];
    __shared__ __attribute__((aligned(32))) __bf16 lAi[2][BM][LDP];
    __shared__ __attribute__((aligned(32))) __bf16 lBr[2][BN][LDP];   // transposed
    __shared__ __attribute__((aligned(32))) __bf16 lBi[2][BN][LDP];   // transposed

    const int tid  = threadIdx.x;
    const int wv   = tid >> 5;
    const int lane = tid & 31;
    const int hw   = lane >> 4;
    const int r16  = lane & 15;
    const int mBase = blockIdx.y * BM;
    const int nBase = blockIdx.x * BN;

    v8f accR[4] = {};
    v8f accI[4] = {};

    float2 sar[8], sai[8];
    v8bf   bar[2], bai[2];
    float4 sbr[4], sbi[4];

    auto gload = [&](int k0) {
        if constexpr (sizeof(TA) == 4) {
            #pragma unroll
            for (int i = 0; i < 8; ++i) {               // 1024 float2 slots
                int idx = tid + i * 128;
                int row = idx >> 4, c = (idx & 15) << 1;
                int k = k0 + c;
                float2 vr = {0.f, 0.f}, vi = {0.f, 0.f};
                if (k < K) {                             // K even -> pair safe
                    vr = *(const float2*)((const float*)Ar + (size_t)(mBase + row) * lda + k);
                    vi = *(const float2*)((const float*)Ai + (size_t)(mBase + row) * lda + k);
                }
                sar[i] = vr; sai[i] = vi;
            }
        } else {
            #pragma unroll
            for (int i = 0; i < 2; ++i) {               // 256 chunks of 8 bf16
                int idx = tid + i * 128;
                int row = idx >> 2, c = (idx & 3) << 3;
                int k = k0 + c;
                v8bf vr = {}, vi = {};
                if (k < K) {
                    vr = *(const v8bf*)((const __bf16*)Ar + (size_t)(mBase + row) * lda + k);
                    vi = *(const v8bf*)((const __bf16*)Ai + (size_t)(mBase + row) * lda + k);
                }
                bar[i] = vr; bai[i] = vi;
            }
        }
        #pragma unroll
        for (int i = 0; i < 4; ++i) {
            int idx = tid + i * 128;
            int row = idx >> 4, c4 = (idx & 15) << 2;
            int k = k0 + row;
            float4 vr = {0.f, 0.f, 0.f, 0.f}, vi = {0.f, 0.f, 0.f, 0.f};
            if (k < K) {
                vr = *(const float4*)(Br + (size_t)k * ldb + nBase + c4);
                vi = *(const float4*)(Bi + (size_t)k * ldb + nBase + c4);
            }
            sbr[i] = vr; sbi[i] = vi;
        }
    };

    auto sstore = [&](int buf) {
        if constexpr (sizeof(TA) == 4) {
            #pragma unroll
            for (int i = 0; i < 8; ++i) {
                int idx = tid + i * 128;
                int row = idx >> 4, c = (idx & 15) << 1;
                int p = c;                               // contiguous k order
                lAr[buf][row][p + 0] = (__bf16)sar[i].x;
                lAr[buf][row][p + 1] = (__bf16)sar[i].y;
                lAi[buf][row][p + 0] = (__bf16)sai[i].x;
                lAi[buf][row][p + 1] = (__bf16)sai[i].y;
            }
        } else {
            #pragma unroll
            for (int i = 0; i < 2; ++i) {
                int idx = tid + i * 128;
                int row = idx >> 2, c = (idx & 3) << 3;
                int p = c;                               // contiguous k order
                *(v8bf*)&lAr[buf][row][p] = bar[i];
                *(v8bf*)&lAi[buf][row][p] = bai[i];
            }
        }
        #pragma unroll
        for (int i = 0; i < 4; ++i) {
            int idx = tid + i * 128;
            int k = idx >> 4, c = (idx & 15) << 2;
            lBr[buf][c + 0][k] = (__bf16)sbr[i].x;
            lBr[buf][c + 1][k] = (__bf16)sbr[i].y;
            lBr[buf][c + 2][k] = (__bf16)sbr[i].z;
            lBr[buf][c + 3][k] = (__bf16)sbr[i].w;
            lBi[buf][c + 0][k] = (__bf16)sbi[i].x;
            lBi[buf][c + 1][k] = (__bf16)sbi[i].y;
            lBi[buf][c + 2][k] = (__bf16)sbi[i].z;
            lBi[buf][c + 3][k] = (__bf16)sbi[i].w;
        }
    };

    gload(0);
    sstore(0);
    const int nk = (K + BK - 1) / BK;
    for (int kt = 0; kt < nk; ++kt) {
        __syncthreads();
        if (kt + 1 < nk) gload((kt + 1) * BK);
        if (kt + 2 < nk) {
            __builtin_prefetch((const char*)Ar + (size_t)(mBase + (tid & 63)) * lda * sizeof(TA)
                                               + (size_t)((kt + 2) * BK) * sizeof(TA), 0, 3);
            __builtin_prefetch(Br + (size_t)((kt + 2) * BK + (tid & 31)) * ldb + nBase, 0, 3);
        }
        const int buf = kt & 1;
        const v16bf arf = *(const v16bf*)&lAr[buf][wv * 16 + r16][hw * 16];
        const v16bf aif = *(const v16bf*)&lAi[buf][wv * 16 + r16][hw * 16];
        const v16bf anf = negbf(aif);                   // 8x v_xor_b32
        #pragma unroll
        for (int nt = 0; nt < 4; ++nt) {
            const v16bf brf = *(const v16bf*)&lBr[buf][nt * 16 + r16][hw * 16];
            const v16bf bif = *(const v16bf*)&lBi[buf][nt * 16 + r16][hw * 16];
            accR[nt] = __builtin_amdgcn_wmma_f32_16x16x32_bf16(
                false, arf, false, brf, (short)0, accR[nt], false, false);
            accR[nt] = __builtin_amdgcn_wmma_f32_16x16x32_bf16(
                false, anf, false, bif, (short)0, accR[nt], false, false);
            accI[nt] = __builtin_amdgcn_wmma_f32_16x16x32_bf16(
                false, arf, false, bif, (short)0, accI[nt], false, false);
            accI[nt] = __builtin_amdgcn_wmma_f32_16x16x32_bf16(
                false, aif, false, brf, (short)0, accI[nt], false, false);
        }
        if (kt + 1 < nk) sstore((kt + 1) & 1);
    }

    #pragma unroll
    for (int nt = 0; nt < 4; ++nt) {
        int n = nBase + nt * 16 + r16;
        float br = biasR ? biasR[n] : 0.f;
        float bi = biasI ? biasI[n] : 0.f;
        #pragma unroll
        for (int i = 0; i < 8; ++i) {
            int m = mBase + wv * 16 + hw * 8 + i;
            Cr[(size_t)m * ldc + colOff + n] = (TO)(accR[nt][i] + br);
            Ci[(size_t)m * ldc + colOff + n] = (TO)(accI[nt][i] + bi);
        }
    }
}

// ---------------------------------------------------------------------------
// Kernel 6: gate GEMM with virtual concat A = [x | frf | fif | mag], fused
// sigmoid + mix epilogue.  N = H = 1024, K = 4H = 4096.  128x64 tile.
// ---------------------------------------------------------------------------
__global__ __launch_bounds__(128) void gate_kernel(
    const float* __restrict__ x,
    const float* __restrict__ frf, const float* __restrict__ fif,
    const float* __restrict__ gW, const float* __restrict__ gb,
    float* __restrict__ out)
{
    constexpr int Hh = 1024;
    constexpr int K  = 4096;

    __shared__ __attribute__((aligned(32))) __bf16 lA [2][BMR][LDP];
    __shared__ __attribute__((aligned(32))) __bf16 lBt[2][BN][LDP];

    const int tid  = threadIdx.x;
    const int wv   = tid >> 5;
    const int lane = tid & 31;
    const int hw   = lane >> 4;
    const int r16  = lane & 15;
    const int mBase = blockIdx.y * BMR;
    const int nBase = blockIdx.x * BN;

    v8f acc[2][4] = {};
    float4 ra[8], rb[4];

    auto gload = [&](int k0) {
        #pragma unroll
        for (int i = 0; i < 8; ++i) {
            int idx = tid + i * 128;
            int row = idx >> 3, c = (idx & 7) << 2;
            int k = k0 + c;
            int m = mBase + row;
            float4 v;
            if (k < Hh) {
                v = *(const float4*)(x + (size_t)m * Hh + k);
            } else if (k < 2 * Hh) {
                v = *(const float4*)(frf + (size_t)m * Hh + (k - Hh));
            } else if (k < 3 * Hh) {
                v = *(const float4*)(fif + (size_t)m * Hh + (k - 2 * Hh));
            } else {
                const float4 a = *(const float4*)(frf + (size_t)m * Hh + (k - 3 * Hh));
                const float4 b = *(const float4*)(fif + (size_t)m * Hh + (k - 3 * Hh));
                v.x = sqrtf(a.x * a.x + b.x * b.x + EPS_C);
                v.y = sqrtf(a.y * a.y + b.y * b.y + EPS_C);
                v.z = sqrtf(a.z * a.z + b.z * b.z + EPS_C);
                v.w = sqrtf(a.w * a.w + b.w * b.w + EPS_C);
            }
            ra[i] = v;
        }
        #pragma unroll
        for (int i = 0; i < 4; ++i) {
            int idx = tid + i * 128;
            int row = idx >> 4, c4 = (idx & 15) << 2;
            rb[i] = *(const float4*)(gW + (size_t)(k0 + row) * Hh + nBase + c4);
        }
    };
    auto sstore = [&](int buf) {
        #pragma unroll
        for (int i = 0; i < 8; ++i) {
            int idx = tid + i * 128;
            int row = idx >> 3, c = (idx & 7) << 2;
            __bf16* d = &lA[buf][row][c];
            d[0] = (__bf16)ra[i].x; d[1] = (__bf16)ra[i].y;
            d[2] = (__bf16)ra[i].z; d[3] = (__bf16)ra[i].w;
        }
        #pragma unroll
        for (int i = 0; i < 4; ++i) {
            int idx = tid + i * 128;
            int k = idx >> 4, c = (idx & 15) << 2;
            lBt[buf][c + 0][k] = (__bf16)rb[i].x;
            lBt[buf][c + 1][k] = (__bf16)rb[i].y;
            lBt[buf][c + 2][k] = (__bf16)rb[i].z;
            lBt[buf][c + 3][k] = (__bf16)rb[i].w;
        }
    };

    gload(0);
    sstore(0);
    const int nk = K / BK;
    for (int kt = 0; kt < nk; ++kt) {
        __syncthreads();
        if (kt + 1 < nk) gload((kt + 1) * BK);
        if (kt + 2 < nk) {
            __builtin_prefetch(gW + (size_t)((kt + 2) * BK + (tid & 31)) * Hh + nBase, 0, 3);
        }
        const int buf = kt & 1;
        const v16bf af0 = *(const v16bf*)&lA[buf][wv * 32 + r16][hw * 16];
        const v16bf af1 = *(const v16bf*)&lA[buf][wv * 32 + 16 + r16][hw * 16];
        #pragma unroll
        for (int nt = 0; nt < 4; ++nt) {
            const v16bf bf = *(const v16bf*)&lBt[buf][nt * 16 + r16][hw * 16];
            acc[0][nt] = __builtin_amdgcn_wmma_f32_16x16x32_bf16(
                false, af0, false, bf, (short)0, acc[0][nt], false, false);
            acc[1][nt] = __builtin_amdgcn_wmma_f32_16x16x32_bf16(
                false, af1, false, bf, (short)0, acc[1][nt], false, false);
        }
        if (kt + 1 < nk) sstore((kt + 1) & 1);
    }

    #pragma unroll
    for (int g = 0; g < 2; ++g) {
        #pragma unroll
        for (int nt = 0; nt < 4; ++nt) {
            int n = nBase + nt * 16 + r16;
            float bv = gb[n];
            #pragma unroll
            for (int i = 0; i < 8; ++i) {
                int m = mBase + wv * 32 + g * 16 + hw * 8 + i;
                float v = acc[g][nt][i] + bv;
                float gv = 1.f / (1.f + __expf(-v));
                float f = frf[(size_t)m * Hh + n];
                float xv = x[(size_t)m * Hh + n];
                out[(size_t)m * Hh + n] = gv * f + (1.f - gv) * xv;
            }
        }
    }
}

// ---------------------------------------------------------------------------
// Host launch
// ---------------------------------------------------------------------------
extern "C" void kernel_launch(void* const* d_in, const int* in_sizes, int n_in,
                              void* d_out, int out_size, void* d_ws, size_t ws_size,
                              hipStream_t stream)
{
    const float* x      = (const float*)d_in[0];
    const float* sig_Wr = (const float*)d_in[1];
    const float* sig_Wi = (const float*)d_in[2];
    const float* sig_br = (const float*)d_in[3];
    const float* sig_bi = (const float*)d_in[4];
    const float* A_r    = (const float*)d_in[5];
    const float* A_i    = (const float*)d_in[6];
    const float* lam_r  = (const float*)d_in[7];
    const float* lam_i  = (const float*)d_in[8];
    const float* eh_Wr  = (const float*)d_in[9];
    const float* eh_Wi  = (const float*)d_in[10];
    const float* eh_br  = (const float*)d_in[11];
    const float* eh_bi  = (const float*)d_in[12];
    const float* fu_Wr  = (const float*)d_in[13];
    const float* fu_Wi  = (const float*)d_in[14];
    const float* fu_br  = (const float*)d_in[15];
    const float* fu_bi  = (const float*)d_in[16];
    const float* g_W    = (const float*)d_in[17];
    const float* g_b    = (const float*)d_in[18];

    const int Mrows = 8192;   // B*T

    char* ws = (char*)d_ws;
    size_t off = 0;
    auto alloc = [&](size_t bytes) -> void* {
        void* p = ws + off;
        off = (off + bytes + 255) & ~(size_t)255;
        return p;
    };

    float*  zr  = (float*)alloc((size_t)Mrows * 512 * 4);
    float*  zi  = (float*)alloc((size_t)Mrows * 512 * 4);
    float*  ur  = (float*)alloc((size_t)4 * Mrows * 16 * 4);
    float*  ui  = (float*)alloc((size_t)4 * Mrows * 16 * 4);
    float*  fr  = (float*)alloc((size_t)4 * Mrows * 38 * 4);
    float*  fi  = (float*)alloc((size_t)4 * Mrows * 38 * 4);
    __bf16* yrc = (__bf16*)alloc((size_t)Mrows * 4096 * 2);
    __bf16* yic = (__bf16*)alloc((size_t)Mrows * 4096 * 2);
    float*  frf = (float*)alloc((size_t)Mrows * 1024 * 4);
    float*  fif = (float*)alloc((size_t)Mrows * 1024 * 4);

    const dim3 blk(128);

    // 1) sigma projection: zr, zi
    gemm_rr_kernel<<<dim3(512 / BN, Mrows / BMR), blk, 0, stream>>>(
        x, 1024, sig_Wr, 512, sig_br, zr, 512, Mrows, 512, 1024);
    gemm_rr_kernel<<<dim3(512 / BN, Mrows / BMR), blk, 0, stream>>>(
        x, 1024, sig_Wi, 512, sig_bi, zi, 512, Mrows, 512, 1024);

    // 2) engine drive projection (TDM-staged LDS)
    uproj_kernel<<<Mrows, blk, 0, stream>>>(zr, zi, A_r, A_i, ur, ui);

    // 3) sequential recurrence (16 chains, one wave each)
    engine_kernel<<<16, 32, 0, stream>>>(ur, ui, lam_r, lam_i, fr, fi);

    // 4) engine head: complex GEMM per engine, K=38, output bf16 into [M, 4096]
    for (int e = 0; e < 4; ++e) {
        cgemm_kernel<float, __bf16><<<dim3(1024 / BN, Mrows / BM), blk, 0, stream>>>(
            fr + (size_t)e * Mrows * 38, fi + (size_t)e * Mrows * 38, 38,
            eh_Wr + (size_t)e * 38 * 1024, eh_Wi + (size_t)e * 38 * 1024, 1024,
            eh_br + (size_t)e * 1024, eh_bi + (size_t)e * 1024,
            yrc, yic, 4096, e * 1024, Mrows, 1024, 38);
    }

    // 5) fuse: complex GEMM, bf16 A, K=4096
    cgemm_kernel<__bf16, float><<<dim3(1024 / BN, Mrows / BM), blk, 0, stream>>>(
        yrc, yic, 4096, fu_Wr, fu_Wi, 1024, fu_br, fu_bi,
        frf, fif, 1024, 0, Mrows, 1024, 4096);

    // 6) gate + mix (fused epilogue), writes final output
    gate_kernel<<<dim3(1024 / BN, Mrows / BMR), blk, 0, stream>>>(
        x, frf, fif, g_W, g_b, (float*)d_out);
}
